// KoLeoLoss_8572754722868
// MI455X (gfx1250) — matrix operand order repaired and checked by reading, not compiled, not run on hardware
//
#include <hip/hip_runtime.h>
#include <hip/hip_bf16.h>
#include <math.h>

// ---------------------------------------------------------------------------
// KoLeo loss for [16384, 1024] fp32:
//   1) row-normalize -> bf16 copy (WMMA feed) + fp32 norms
//   2) 1-NN via tiled Gram matrix using v_wmma_f32_16x16x32_bf16
//      2 A-frags x 4 B-frags per wave (32x64 tile) -> 1.5 loads per WMMA
//   3) exact fp32 pairwise distance to the NN, log
//   4) -mean reduction
// ---------------------------------------------------------------------------

typedef __attribute__((ext_vector_type(16))) __bf16 v16bf;
typedef __attribute__((ext_vector_type(8)))  __bf16 v8bf;
typedef __attribute__((ext_vector_type(8)))  float  v8f;

constexpr int   BN   = 16384;   // rows
constexpr int   DD   = 1024;    // cols
constexpr float KEPS = 1e-8f;

constexpr int ROWS_PER_WAVE = 32;  // 2 A fragments of 16 rows
constexpr int WAVES_PER_WG  = 4;   // 128-thread blocks
constexpr int ROWS_PER_WG   = ROWS_PER_WAVE * WAVES_PER_WG;  // 128
constexpr int JCHUNK        = 64;  // 4 WMMA tiles of 16 columns

// ---------------------------------------------------------------------------
// Pass 1: row L2 norms; write bf16 normalized matrix + fp32 norms
// ---------------------------------------------------------------------------
__global__ __launch_bounds__(256) void koleo_normalize_kernel(
    const float* __restrict__ x, __bf16* __restrict__ xn, float* __restrict__ norms)
{
    const int row = blockIdx.x;
    const float* xr = x + (size_t)row * DD;

    float ss = 0.f;
    for (int k = threadIdx.x; k < DD; k += 256) {
        float v = xr[k];
        ss += v * v;
    }
    #pragma unroll
    for (int s = 1; s < 32; s <<= 1) ss += __shfl_xor(ss, s, 32);

    __shared__ float wsum[8];
    if ((threadIdx.x & 31) == 0) wsum[threadIdx.x >> 5] = ss;
    __syncthreads();

    float tot = 0.f;
    #pragma unroll
    for (int w = 0; w < 8; ++w) tot += wsum[w];

    const float nrm = sqrtf(tot);
    const float inv = 1.0f / fmaxf(nrm, KEPS);
    if (threadIdx.x == 0) norms[row] = nrm;

    __bf16* xo = xn + (size_t)row * DD;
    for (int k = threadIdx.x; k < DD; k += 256)
        xo[k] = (__bf16)(xr[k] * inv);
}

// ---------------------------------------------------------------------------
// Pass 2: 1-NN index per row via WMMA Gram tiles.
// Each wave owns 32 rows (2 A fragments) and sweeps all columns in chunks of
// 64 (4 B fragments, 8 f32 accumulators), K accumulated in steps of 32.
// Per-lane running (bestv, besti) -> cross-lane reduce only once at the end.
// ---------------------------------------------------------------------------
__global__ __launch_bounds__(128) void koleo_nn_kernel(
    const __bf16* __restrict__ xn, int* __restrict__ nn_idx)
{
    const int lane  = threadIdx.x & 31;
    const int wave  = threadIdx.x >> 5;
    const int i0    = blockIdx.x * ROWS_PER_WG + wave * ROWS_PER_WAVE;
    const int m     = lane & 15;   // row (A) / col (B) within tile
    const int khalf = lane >> 4;   // K-half select per ISA VGPR layout

    const __bf16* arow0 = xn + (size_t)(i0 + m) * DD;        // rows i0..i0+15
    const __bf16* arow1 = arow0 + (size_t)16 * DD;           // rows i0+16..i0+31

    // per-lane running best for rows (r + 8*khalf) [+16 for frag 1],
    // restricted to columns == m (mod 16)
    float bv0[8], bv1[8];
    int   bi0[8], bi1[8];
    #pragma unroll
    for (int r = 0; r < 8; ++r) {
        bv0[r] = -3.0f; bi0[r] = 0;
        bv1[r] = -3.0f; bi1[r] = 0;
    }

    for (int j0 = 0; j0 < BN; j0 += JCHUNK) {
        v8f c0[4] = {};
        v8f c1[4] = {};

        for (int kb = 0; kb < DD; kb += 32) {
            // --- A fragments (16x32 bf16): lane m holds its row.
            //     lanes 0-15: K = kb+0..7 (v0-3), kb+16..23 (v4-7)
            //     lanes 16-31: K = kb+8..15,      kb+24..31
            const v8bf* pa0 = reinterpret_cast<const v8bf*>(arow0 + kb + khalf * 8);
            const v8bf* pa1 = reinterpret_cast<const v8bf*>(arow1 + kb + khalf * 8);
            v16bf a0 = __builtin_shufflevector(pa0[0], pa0[2],
                0,1,2,3,4,5,6,7,8,9,10,11,12,13,14,15);
            v16bf a1 = __builtin_shufflevector(pa1[0], pa1[2],
                0,1,2,3,4,5,6,7,8,9,10,11,12,13,14,15);

            // --- B fragments (32x16 bf16): lane m holds column j,
            //     lanes 0-15: K = kb+0..7, kb+16..23 ; lanes 16-31: K = kb+8..15, kb+24..31
            #pragma unroll
            for (int t = 0; t < 4; ++t) {
                const __bf16* brow = xn + (size_t)(j0 + t * 16 + m) * DD + kb + khalf * 8;
                const v8bf* pb = reinterpret_cast<const v8bf*>(brow);
                v16bf bfrag = __builtin_shufflevector(pb[0], pb[2],
                    0,1,2,3,4,5,6,7,8,9,10,11,12,13,14,15);

                c0[t] = __builtin_amdgcn_wmma_f32_16x16x32_bf16(
                    false, a0, false, bfrag, (short)0, c0[t], false, false);
                c1[t] = __builtin_amdgcn_wmma_f32_16x16x32_bf16(
                    false, a1, false, bfrag, (short)0, c1[t], false, false);
            }
        }

        // --- branchless per-lane argmax update (pure VALU, no cross-lane ops)
        #pragma unroll
        for (int t = 0; t < 4; ++t) {
            const int colg = j0 + t * 16 + m;  // this lane's column
            #pragma unroll
            for (int r = 0; r < 8; ++r) {
                // C layout: VGPR r, lane l -> row = r + 8*(l>=16), col = l%16
                const int rowg0 = i0 + r + khalf * 8;
                const int rowg1 = rowg0 + 16;

                float v0 = c0[t][r];
                v0 = (rowg0 == colg) ? -1.0f : v0;  // reference: diagonal := -1
                const bool take0 = (v0 > bv0[r]) | ((v0 == bv0[r]) & (colg < bi0[r]));
                bv0[r] = take0 ? v0   : bv0[r];
                bi0[r] = take0 ? colg : bi0[r];

                float v1 = c1[t][r];
                v1 = (rowg1 == colg) ? -1.0f : v1;
                const bool take1 = (v1 > bv1[r]) | ((v1 == bv1[r]) & (colg < bi1[r]));
                bv1[r] = take1 ? v1   : bv1[r];
                bi1[r] = take1 ? colg : bi1[r];
            }
        }
    }

    // --- one-time cross-lane reduction: 16-lane butterfly + half swap
    int outi0 = 0, outi1 = 0;  // lane L (0..15): rows i0+L and i0+16+L
    #pragma unroll
    for (int r = 0; r < 8; ++r) {
        float v0 = bv0[r]; int id0 = bi0[r];
        float v1 = bv1[r]; int id1 = bi1[r];
        #pragma unroll
        for (int s = 1; s < 16; s <<= 1) {
            const float ov0 = __shfl_xor(v0, s, 32);
            const int   oi0 = __shfl_xor(id0, s, 32);
            const bool  t0  = (ov0 > v0) | ((ov0 == v0) & (oi0 < id0));
            v0  = t0 ? ov0 : v0;
            id0 = t0 ? oi0 : id0;

            const float ov1 = __shfl_xor(v1, s, 32);
            const int   oi1 = __shfl_xor(id1, s, 32);
            const bool  t1  = (ov1 > v1) | ((ov1 == v1) & (oi1 < id1));
            v1  = t1 ? ov1 : v1;
            id1 = t1 ? oi1 : id1;
        }
        // lower half holds row r's result; upper half holds row r+8's.
        const int s0 = __shfl_xor(id0, 16, 32);
        const int s1 = __shfl_xor(id1, 16, 32);
        outi0 = (lane == r)     ? id0 : outi0;
        outi0 = (lane == r + 8) ? s0  : outi0;
        outi1 = (lane == r)     ? id1 : outi1;
        outi1 = (lane == r + 8) ? s1  : outi1;
    }

    if (lane < 16) {
        nn_idx[i0 + lane]      = outi0;
        nn_idx[i0 + 16 + lane] = outi1;
    }
}

// ---------------------------------------------------------------------------
// Pass 3: exact fp32 log-distance per row (matches reference numerics)
// ---------------------------------------------------------------------------
__global__ __launch_bounds__(256) void koleo_dist_kernel(
    const float* __restrict__ x, const float* __restrict__ norms,
    const int* __restrict__ nn_idx, float* __restrict__ logd)
{
    const int row = blockIdx.x;
    const int nb  = nn_idx[row];
    const float invi = 1.0f / fmaxf(norms[row], KEPS);
    const float invj = 1.0f / fmaxf(norms[nb],  KEPS);
    const float* xi = x + (size_t)row * DD;
    const float* xj = x + (size_t)nb  * DD;

    float ss = 0.f;
    for (int k = threadIdx.x; k < DD; k += 256) {
        float d = xi[k] * invi - xj[k] * invj + KEPS;
        ss += d * d;
    }
    #pragma unroll
    for (int s = 1; s < 32; s <<= 1) ss += __shfl_xor(ss, s, 32);

    __shared__ float wsum[8];
    if ((threadIdx.x & 31) == 0) wsum[threadIdx.x >> 5] = ss;
    __syncthreads();

    if (threadIdx.x == 0) {
        float tot = 0.f;
        #pragma unroll
        for (int w = 0; w < 8; ++w) tot += wsum[w];
        logd[row] = logf(sqrtf(tot) + KEPS);
    }
}

// ---------------------------------------------------------------------------
// Pass 4: deterministic single-block -mean (no atomics: graph-replay safe)
// ---------------------------------------------------------------------------
__global__ __launch_bounds__(256) void koleo_final_kernel(
    const float* __restrict__ logd, float* __restrict__ out)
{
    float s = 0.f;
    for (int i = threadIdx.x; i < BN; i += 256) s += logd[i];
    #pragma unroll
    for (int sh = 1; sh < 32; sh <<= 1) s += __shfl_xor(s, sh, 32);

    __shared__ float wsum[8];
    if ((threadIdx.x & 31) == 0) wsum[threadIdx.x >> 5] = s;
    __syncthreads();

    if (threadIdx.x == 0) {
        float tot = 0.f;
        #pragma unroll
        for (int w = 0; w < 8; ++w) tot += wsum[w];
        out[0] = -tot / (float)BN;
    }
}

// ---------------------------------------------------------------------------
extern "C" void kernel_launch(void* const* d_in, const int* in_sizes, int n_in,
                              void* d_out, int out_size, void* d_ws, size_t ws_size,
                              hipStream_t stream)
{
    const float* x   = (const float*)d_in[0];
    float*       out = (float*)d_out;

    char* ws = (char*)d_ws;
    const size_t xn_bytes = (size_t)BN * DD * sizeof(__bf16);  // 32 MB
    __bf16* xn    = (__bf16*)ws;
    float*  norms = (float*)(ws + xn_bytes);
    int*    nnidx = (int*)  (ws + xn_bytes + (size_t)BN * 4);
    float*  logd  = (float*)(ws + xn_bytes + (size_t)BN * 8);

    koleo_normalize_kernel<<<BN, 256, 0, stream>>>(x, xn, norms);
    koleo_nn_kernel<<<BN / ROWS_PER_WG, 128, 0, stream>>>(xn, nnidx);
    koleo_dist_kernel<<<BN, 256, 0, stream>>>(x, norms, nnidx, logd);
    koleo_final_kernel<<<1, 256, 0, stream>>>(logd, out);
}
